// RetrievalTransformerNAR_56461640073576
// MI455X (gfx1250) — matrix operand order repaired and checked by reading, not compiled, not run on hardware
//
#include <hip/hip_runtime.h>
#include <cstdint>

// ---------------------------------------------------------------------------
// RetrievalTransformerNAR pipeline for gfx1250 (MI455X, wave32, WMMA).
// ---------------------------------------------------------------------------

typedef __attribute__((ext_vector_type(16))) _Float16 v16h;
typedef __attribute__((ext_vector_type(8)))  float    v8f;

#define BQ   512
#define DIM  768
#define VSZ  100000
#define KC   80

// ---------------------------------------------------------------------------
// Generic WMMA GEMM:  C[M,N] = act( A[M,K] * B + bias )
//   bmode 0: B is [K,N] row-major (weights)
//   bmode 1: B is [N,K] row-major (B^T; optional per-row scale) — vocab sims
//   act: 0 none, 1 tanh, 2 relu
// Tile: (W*16) M x (NSUB*16) N, blockDim = W*32, K step 32.
// K/lda/ldb even, k0/n0 even -> float2 staging always 8B aligned.
// Fast path (no clamp/mask) for interior tiles; M is always a TM multiple here.
// ---------------------------------------------------------------------------
template<int W, int NSUB>
__global__ __launch_bounds__(W * 32) void gemm_wmma_kernel(
    const float* __restrict__ A, int lda,
    const float* __restrict__ Bm, int ldb,
    const float* __restrict__ bscale,
    const float* __restrict__ bias,
    float* __restrict__ C, int ldc, int coloff,
    int M, int N, int K, int bmode, int act)
{
  constexpr int TM = W * 16;
  constexpr int TN = NSUB * 16;
  constexpr int BD = W * 32;
  constexpr int NA = 8;                  // A float2 per thread (TM*16 / BD)
  constexpr int NB = (TN * 16) / BD;     // B float2 per thread

  __shared__ _Float16 As[TM * 36];       // 36 = 32 + pad
  __shared__ _Float16 Bs[TN * 36];       // staged transposed: row = N column

  const int tid  = threadIdx.x;
  const int wave = tid >> 5;
  const int lane = tid & 31;
  const int m0 = blockIdx.y * TM;
  const int n0 = blockIdx.x * TN;
  const bool mfull = (m0 + TM <= M);
  const bool nfull = (n0 + TN <= N);

  float2 aR[NA];
  float2 bR[NB];

  auto loadA = [&](int k0) {
    if (mfull && (k0 + 32 <= K)) {
      #pragma unroll
      for (int i = 0; i < NA; ++i) {
        int u = tid + i * BD;
        int r = u >> 4, c2 = (u & 15) << 1;
        aR[i] = *(const float2*)(A + (size_t)(m0 + r) * lda + k0 + c2);
      }
    } else {
      #pragma unroll
      for (int i = 0; i < NA; ++i) {
        int u = tid + i * BD;
        int r = u >> 4, c2 = (u & 15) << 1;
        int gm = m0 + r, gk = k0 + c2;
        int cm = gm < M ? gm : M - 1;
        int ck = gk < K ? gk : K - 2;                  // K even -> stays even
        float2 v = *(const float2*)(A + (size_t)cm * lda + ck);
        float msk = (gm < M && gk < K) ? 1.0f : 0.0f;
        aR[i].x = v.x * msk; aR[i].y = v.y * msk;
      }
    }
  };
  auto loadB = [&](int k0) {
    if (bmode == 1) {
      if (nfull && (k0 + 32 <= K)) {
        #pragma unroll
        for (int i = 0; i < NB; ++i) {
          int u = tid + i * BD;
          int n = u >> 4, c2 = (u & 15) << 1;
          int gn = n0 + n;
          float2 v = *(const float2*)(Bm + (size_t)gn * ldb + k0 + c2);
          float sc = bscale ? bscale[gn] : 1.0f;
          bR[i].x = v.x * sc; bR[i].y = v.y * sc;
        }
      } else {
        #pragma unroll
        for (int i = 0; i < NB; ++i) {
          int u = tid + i * BD;
          int n = u >> 4, c2 = (u & 15) << 1;
          int gn = n0 + n, gk = k0 + c2;
          int cn = gn < N ? gn : N - 1;
          int ck = gk < K ? gk : K - 2;
          float2 v = *(const float2*)(Bm + (size_t)cn * ldb + ck);
          float msk = (gn < N && gk < K) ? (bscale ? bscale[cn] : 1.0f) : 0.0f;
          bR[i].x = v.x * msk; bR[i].y = v.y * msk;
        }
      }
    } else {
      if (nfull && (k0 + 32 <= K)) {
        #pragma unroll
        for (int i = 0; i < NB; ++i) {
          int u = tid + i * BD;
          int k = u / (TN / 2), n2 = u % (TN / 2);
          bR[i] = *(const float2*)(Bm + (size_t)(k0 + k) * ldb + n0 + 2 * n2);
        }
      } else {
        #pragma unroll
        for (int i = 0; i < NB; ++i) {
          int u = tid + i * BD;
          int k = u / (TN / 2), n2 = u % (TN / 2);
          int gk = k0 + k, gn = n0 + 2 * n2;
          int ck = gk < K ? gk : K - 1;
          int cn = gn < N ? gn : N - 2;                // N even for weights
          float2 v = *(const float2*)(Bm + (size_t)ck * ldb + cn);
          float ms = (gk < K && gn < N) ? 1.0f : 0.0f;
          bR[i].x = v.x * ms; bR[i].y = v.y * ms;
        }
      }
    }
  };
  auto storeA = [&]() {
    #pragma unroll
    for (int i = 0; i < NA; ++i) {
      int u = tid + i * BD;
      int r = u >> 4, c2 = (u & 15) << 1;
      union { _Float16 h[2]; uint32_t w; } p;
      p.h[0] = (_Float16)aR[i].x; p.h[1] = (_Float16)aR[i].y;
      *(uint32_t*)(As + r * 36 + c2) = p.w;
    }
  };
  auto storeB = [&]() {
    if (bmode == 1) {
      #pragma unroll
      for (int i = 0; i < NB; ++i) {
        int u = tid + i * BD;
        int n = u >> 4, c2 = (u & 15) << 1;
        union { _Float16 h[2]; uint32_t w; } p;
        p.h[0] = (_Float16)bR[i].x; p.h[1] = (_Float16)bR[i].y;
        *(uint32_t*)(Bs + n * 36 + c2) = p.w;
      }
    } else {
      #pragma unroll
      for (int i = 0; i < NB; ++i) {
        int u = tid + i * BD;
        int k = u / (TN / 2), n2 = u % (TN / 2);
        Bs[(2 * n2) * 36 + k]     = (_Float16)bR[i].x;
        Bs[(2 * n2 + 1) * 36 + k] = (_Float16)bR[i].y;
      }
    }
  };

  v8f acc[NSUB] = {};
  loadA(0); loadB(0);

  union Frag { v16h v; uint32_t u[8]; };
  for (int k0 = 0; k0 < K; k0 += 32) {
    storeA(); storeB();
    __syncthreads();
    if (k0 + 32 < K) { loadA(k0 + 32); loadB(k0 + 32); }   // overlap with WMMA

    // A fragment (16x32): ISA pair layout {0-3,8-11} / {4-7,12-15}
    Frag af;
    {
      int row = wave * 16 + (lane & 15);
      int sel = (lane >> 4) << 2;
      const uint32_t* p = (const uint32_t*)(As + row * 36);
      #pragma unroll
      for (int j = 0; j < 8; ++j) {
        int pair = (j & 3) + ((j >> 2) << 3) + sel;
        af.u[j] = p[pair];
      }
    }
    // all B fragments first (lets DS loads pipeline), then WMMAs back-to-back
    Frag bf[NSUB];
    #pragma unroll
    for (int nt = 0; nt < NSUB; ++nt) {
      int row = nt * 16 + (lane & 15);
      int sel = (lane >> 4) << 2;
      const uint32_t* p = (const uint32_t*)(Bs + row * 36);
      #pragma unroll
      for (int j = 0; j < 8; ++j) bf[nt].u[j] = p[(j & 3) + ((j >> 2) << 3) + sel];
    }
    #pragma unroll
    for (int nt = 0; nt < NSUB; ++nt)
      acc[nt] = __builtin_amdgcn_wmma_f32_16x16x32_f16(
          false, af.v, false, bf[nt].v, (short)0, acc[nt], false, false);
    __syncthreads();
  }

  int mrow = m0 + wave * 16 + ((lane >> 4) << 3);
  int ncol = n0 + (lane & 15);
  #pragma unroll
  for (int nt = 0; nt < NSUB; ++nt) {
    int gn = ncol + nt * 16;
    #pragma unroll
    for (int r = 0; r < 8; ++r) {
      int gm = mrow + r;
      if (gm < M && gn < N) {
        float v = acc[nt][r];
        if (bias) v += bias[gn];
        if (act == 1)      v = tanhf(v);
        else if (act == 2) v = fmaxf(v, 0.0f);
        C[(size_t)gm * ldc + coloff + gn] = v;
      }
    }
  }
}

// ---------------------------------------------------------------------------
// comb = concat(neg.mean, assas, neut.mean, pos.mean)   [B,3072]
// ---------------------------------------------------------------------------
__global__ __launch_bounds__(256) void pool_concat_kernel(
    const float* __restrict__ pos, const float* __restrict__ neg,
    const float* __restrict__ neut, const float* __restrict__ assas,
    float* __restrict__ comb)
{
  int b = blockIdx.x;
  for (int d = threadIdx.x; d < DIM; d += 256) {
    float s = 0.0f;
    for (int n = 0; n < 9; ++n) s += neg[((size_t)b * 9 + n) * DIM + d];
    comb[(size_t)b * 3072 + d] = s * (1.0f / 9.0f);
    comb[(size_t)b * 3072 + 768 + d] = assas[(size_t)b * DIM + d];
    s = 0.0f;
    for (int n = 0; n < 6; ++n) s += neut[((size_t)b * 6 + n) * DIM + d];
    comb[(size_t)b * 3072 + 1536 + d] = s * (1.0f / 6.0f);
    s = 0.0f;
    for (int n = 0; n < 9; ++n) s += pos[((size_t)b * 9 + n) * DIM + d];
    comb[(size_t)b * 3072 + 2304 + d] = s * (1.0f / 9.0f);
  }
}

// wave-per-row l2 normalize (768 cols)
__global__ __launch_bounds__(256) void rownorm_kernel(
    const float* __restrict__ src, float* __restrict__ dst, int rows)
{
  int row  = blockIdx.x * 8 + (threadIdx.x >> 5);
  int lane = threadIdx.x & 31;
  if (row >= rows) return;
  const float* s = src + (size_t)row * DIM;
  float v[24]; float ss = 0.0f;
  #pragma unroll
  for (int j = 0; j < 24; ++j) { v[j] = s[lane + j * 32]; ss += v[j] * v[j]; }
  #pragma unroll
  for (int o = 16; o > 0; o >>= 1) ss += __shfl_xor(ss, o);
  float rn = 1.0f / fmaxf(sqrtf(ss), 1e-12f);
  float* d = dst + (size_t)row * DIM;
  #pragma unroll
  for (int j = 0; j < 24; ++j) d[lane + j * 32] = v[j] * rn;
}

// pooled = l2n(mean over 3 heads)
__global__ __launch_bounds__(256) void pooled_kernel(
    const float* __restrict__ heads, float* __restrict__ pooled)
{
  int b    = blockIdx.x * 8 + (threadIdx.x >> 5);
  int lane = threadIdx.x & 31;
  if (b >= BQ) return;
  const float* h = heads + (size_t)b * 3 * DIM;
  float v[24]; float ss = 0.0f;
  #pragma unroll
  for (int j = 0; j < 24; ++j) {
    int d = lane + j * 32;
    float m = (h[d] + h[DIM + d] + h[2 * DIM + d]) * (1.0f / 3.0f);
    v[j] = m; ss += m * m;
  }
  #pragma unroll
  for (int o = 16; o > 0; o >>= 1) ss += __shfl_xor(ss, o);
  float rn = 1.0f / fmaxf(sqrtf(ss), 1e-12f);
  float* p = pooled + (size_t)b * DIM;
  #pragma unroll
  for (int j = 0; j < 24; ++j) p[lane + j * 32] = v[j] * rn;
}

// per-vocab-row inverse norm
__global__ __launch_bounds__(256) void vrnorm_kernel(
    const float* __restrict__ vocab, float* __restrict__ rnorm)
{
  int row  = blockIdx.x * 8 + (threadIdx.x >> 5);
  int lane = threadIdx.x & 31;
  if (row >= VSZ) return;
  const float* s = vocab + (size_t)row * DIM;
  float ss = 0.0f;
  #pragma unroll
  for (int j = 0; j < 24; ++j) { float x = s[lane + j * 32]; ss += x * x; }
  #pragma unroll
  for (int o = 16; o > 0; o >>= 1) ss += __shfl_xor(ss, o);
  if (lane == 0) rnorm[row] = 1.0f / fmaxf(sqrtf(ss), 1e-12f);
}

// ---------------------------------------------------------------------------
// Histogram-based top-k per row (values are cosines in [-1,1]):
//   pass 1: 1024-bin LDS histogram; find threshold bin (cum from top >= k)
//   pass 2: collect candidates >= threshold bin; select k by iterated argmax.
// Deterministic (ties: higher value first, then lower index). sims untouched.
// ---------------------------------------------------------------------------
#define TK_CAP 4096
__global__ __launch_bounds__(256) void topk_hist_kernel(
    const float* __restrict__ s, int n, int k, int* __restrict__ idx_out, int ld)
{
  __shared__ int   hist[1024];
  __shared__ float cval[TK_CAP];
  __shared__ int   cidx[TK_CAP];
  __shared__ float bm[256];
  __shared__ int   bi[256];
  __shared__ int   bslot[256];
  __shared__ int   cnt;
  __shared__ int   tbin;

  int b = blockIdx.x, tid = threadIdx.x;
  const float* row = s + (size_t)b * ld;

  for (int i = tid; i < 1024; i += 256) hist[i] = 0;
  if (tid == 0) cnt = 0;
  __syncthreads();
  for (int j = tid; j < n; j += 256) {
    float v = row[j];
    int bin = (int)((v + 1.0f) * 512.0f);
    bin = bin < 0 ? 0 : (bin > 1023 ? 1023 : bin);
    atomicAdd(&hist[bin], 1);
  }
  __syncthreads();
  if (tid == 0) {
    int cum = 0, t = 0;
    for (int bin = 1023; bin >= 0; --bin) {
      cum += hist[bin];
      if (cum >= k) { t = bin; break; }
    }
    tbin = t;
  }
  __syncthreads();
  int t = tbin;
  for (int j = tid; j < n; j += 256) {
    float v = row[j];
    int bin = (int)((v + 1.0f) * 512.0f);
    bin = bin < 0 ? 0 : (bin > 1023 ? 1023 : bin);
    if (bin >= t) {
      int pos = atomicAdd(&cnt, 1);
      if (pos < TK_CAP) { cval[pos] = v; cidx[pos] = j; }
    }
  }
  __syncthreads();
  int m = cnt < TK_CAP ? cnt : TK_CAP;
  for (int it = 0; it < k; ++it) {
    float best = -INFINITY; int bidx = 0x7fffffff;
    for (int j = tid; j < m; j += 256) {
      float v = cval[j]; int id = cidx[j];
      if (v > best || (v == best && id < bidx)) { best = v; bidx = j; }
    }
    bm[tid] = best;
    bi[tid] = (bidx == 0x7fffffff) ? 0x7fffffff : cidx[bidx];
    bslot[tid] = bidx;
    __syncthreads();
    for (int st = 128; st > 0; st >>= 1) {
      if (tid < st) {
        float v2 = bm[tid + st]; int i2 = bi[tid + st];
        if (v2 > bm[tid] || (v2 == bm[tid] && i2 < bi[tid])) {
          bm[tid] = v2; bi[tid] = i2; bslot[tid] = bslot[tid + st];
        }
      }
      __syncthreads();
    }
    if (tid == 0) {
      idx_out[(size_t)b * k + it] = bi[0];
      if (bslot[0] >= 0 && bslot[0] < TK_CAP) cval[bslot[0]] = -INFINITY;
    }
    __syncthreads();
  }
}

// ---------------------------------------------------------------------------
// Cb[b] = sum l2n(pos) + sum l2n(neut) - 10*l2n(assas)   (NEG_W == 0)
// ---------------------------------------------------------------------------
__global__ __launch_bounds__(256) void combo_kernel(
    const float* __restrict__ pos, const float* __restrict__ neut,
    const float* __restrict__ assas, float* __restrict__ Cb)
{
  __shared__ float red[256];
  int b = blockIdx.x, tid = threadIdx.x;
  float acc[3] = {0.0f, 0.0f, 0.0f};
  auto addvec = [&](const float* vec, float w) {
    float lv[3]; float ss = 0.0f;
    #pragma unroll
    for (int i = 0; i < 3; ++i) { lv[i] = vec[tid + i * 256]; ss += lv[i] * lv[i]; }
    red[tid] = ss; __syncthreads();
    for (int st = 128; st > 0; st >>= 1) { if (tid < st) red[tid] += red[tid + st]; __syncthreads(); }
    float rn = w / fmaxf(sqrtf(red[0]), 1e-12f);
    __syncthreads();
    #pragma unroll
    for (int i = 0; i < 3; ++i) acc[i] += lv[i] * rn;
  };
  for (int n = 0; n < 9; ++n) addvec(pos  + ((size_t)b * 9 + n) * DIM,  1.0f);
  for (int n = 0; n < 6; ++n) addvec(neut + ((size_t)b * 6 + n) * DIM,  1.0f);
  addvec(assas + (size_t)b * DIM, -10.0f);
  #pragma unroll
  for (int i = 0; i < 3; ++i) Cb[(size_t)b * DIM + tid + i * 256] = acc[i];
}

// scores[b,k] = rnorm[idx] * dot(vocab[idx], Cb[b])
__global__ __launch_bounds__(256) void scores_kernel(
    const float* __restrict__ vocab, const float* __restrict__ rnorm,
    const float* __restrict__ Cb, const int* __restrict__ idx80,
    float* __restrict__ sc80)
{
  int b = blockIdx.x, wave = threadIdx.x >> 5, lane = threadIdx.x & 31;
  const float* c = Cb + (size_t)b * DIM;
  for (int k = wave; k < KC; k += 8) {
    int id = idx80[(size_t)b * KC + k];
    const float* w = vocab + (size_t)id * DIM;
    float s = 0.0f;
    for (int d = lane; d < DIM; d += 32) s += w[d] * c[d];
    #pragma unroll
    for (int o = 16; o > 0; o >>= 1) s += __shfl_xor(s, o);
    if (lane == 0) sc80[(size_t)b * KC + k] = s * rnorm[id];
  }
}

// v_in[row] = vocab[idx80[row]] * sc80[row]   (rows = B*80)
__global__ __launch_bounds__(256) void vgather_kernel(
    const float* __restrict__ vocab, const int* __restrict__ idx80,
    const float* __restrict__ sc80, float* __restrict__ v_in)
{
  int r = blockIdx.x;
  int id = idx80[r]; float sc = sc80[r];
  const float* w = vocab + (size_t)id * DIM;
  float* o = v_in + (size_t)r * DIM;
  for (int d = threadIdx.x; d < DIM; d += 256) o[d] = w[d] * sc;
}

// ---------------------------------------------------------------------------
// attention + residual + l2norm
// ---------------------------------------------------------------------------
__global__ __launch_bounds__(256) void attn_kernel(
    const float* __restrict__ queries, const float* __restrict__ vocab,
    const int* __restrict__ idx80, const float* __restrict__ values,
    const float* __restrict__ pooled, float* __restrict__ x)
{
  __shared__ float logits[240];
  __shared__ float wsum[KC];
  __shared__ float red[256];
  int b = blockIdx.x, tid = threadIdx.x, wave = tid >> 5, lane = tid & 31;

  for (int p = wave; p < 240; p += 8) {
    int h = p / KC, k = p - h * KC;
    const float* q = queries + ((size_t)b * 3 + h) * DIM;
    const float* w = vocab + (size_t)idx80[(size_t)b * KC + k] * DIM;
    float s = 0.0f;
    for (int d = lane; d < DIM; d += 32) s += q[d] * w[d];
    #pragma unroll
    for (int o = 16; o > 0; o >>= 1) s += __shfl_xor(s, o);
    if (lane == 0) logits[p] = s;
  }
  __syncthreads();
  if (tid < 3) {
    float mx = -INFINITY;
    for (int k = 0; k < KC; ++k) mx = fmaxf(mx, logits[tid * KC + k]);
    float sm = 0.0f;
    for (int k = 0; k < KC; ++k) { float e = __expf(logits[tid * KC + k] - mx); logits[tid * KC + k] = e; sm += e; }
    float inv = 1.0f / sm;
    for (int k = 0; k < KC; ++k) logits[tid * KC + k] *= inv;
  }
  __syncthreads();
  if (tid < KC) wsum[tid] = logits[tid] + logits[KC + tid] + logits[2 * KC + tid];
  __syncthreads();

  float acc[3];
  #pragma unroll
  for (int i = 0; i < 3; ++i) {
    int d = tid + i * 256;
    float a = 0.0f;
    for (int k = 0; k < KC; ++k) a += wsum[k] * values[((size_t)b * KC + k) * DIM + d];
    acc[i] = a + pooled[(size_t)b * DIM + d];
  }
  float ss = acc[0] * acc[0] + acc[1] * acc[1] + acc[2] * acc[2];
  red[tid] = ss; __syncthreads();
  for (int st = 128; st > 0; st >>= 1) { if (tid < st) red[tid] += red[tid + st]; __syncthreads(); }
  float rn = 1.0f / fmaxf(sqrtf(red[0]), 1e-12f);
  #pragma unroll
  for (int i = 0; i < 3; ++i) x[(size_t)b * DIM + tid + i * 256] = acc[i] * rn;
}

// high[b] = mean of word_embs at top-5 of scores
__global__ __launch_bounds__(256) void high_kernel(
    const float* __restrict__ vocab, const int* __restrict__ idx80,
    const float* __restrict__ sc80, float* __restrict__ high)
{
  __shared__ float s[KC]; __shared__ int t5[5];
  int b = blockIdx.x, tid = threadIdx.x;
  if (tid < KC) s[tid] = sc80[(size_t)b * KC + tid];
  __syncthreads();
  if (tid == 0) {
    for (int it = 0; it < 5; ++it) {
      float best = -INFINITY; int am = 0;
      for (int k = 0; k < KC; ++k) if (s[k] > best) { best = s[k]; am = k; }
      t5[it] = am; s[am] = -INFINITY;
    }
  }
  __syncthreads();
  for (int d = tid; d < DIM; d += 256) {
    float a = 0.0f;
    #pragma unroll
    for (int i = 0; i < 5; ++i)
      a += vocab[(size_t)idx80[(size_t)b * KC + t5[i]] * DIM + d];
    high[(size_t)b * DIM + d] = a * 0.2f;
  }
}

// search_embs gather: rows = B*5
__global__ __launch_bounds__(256) void sgather_kernel(
    const float* __restrict__ vocab, const int* __restrict__ idx5,
    float* __restrict__ se)
{
  int r = blockIdx.x;
  const float* w = vocab + (size_t)idx5[r] * DIM;
  float* o = se + (size_t)r * DIM;
  for (int d = threadIdx.x; d < DIM; d += 256) o[d] = w[d];
}

// ---------------------------------------------------------------------------
extern "C" void kernel_launch(void* const* d_in, const int* in_sizes, int n_in,
                              void* d_out, int out_size, void* d_ws, size_t ws_size,
                              hipStream_t stream)
{
  (void)in_sizes; (void)n_in; (void)out_size; (void)ws_size;
  const float* pos   = (const float*)d_in[0];
  const float* neg   = (const float*)d_in[1];
  const float* neut  = (const float*)d_in[2];
  const float* assas = (const float*)d_in[3];
  const float* eW1 = (const float*)d_in[4];  const float* eb1 = (const float*)d_in[5];
  const float* eW2 = (const float*)d_in[6];  const float* eb2 = (const float*)d_in[7];
  const float* eW3 = (const float*)d_in[8];  const float* eb3 = (const float*)d_in[9];
  const float* eWh = (const float*)d_in[10]; const float* ebh = (const float*)d_in[11];
  const float* vgW1 = (const float*)d_in[12]; const float* vgb1 = (const float*)d_in[13];
  const float* vgW2 = (const float*)d_in[14]; const float* vgb2 = (const float*)d_in[15];
  const float* qgW1 = (const float*)d_in[16]; const float* qgb1 = (const float*)d_in[17];
  const float* qgW2 = (const float*)d_in[18]; const float* qgb2 = (const float*)d_in[19];
  const float* fcW1 = (const float*)d_in[20]; const float* fcb1 = (const float*)d_in[21];
  const float* fcW2 = (const float*)d_in[22]; const float* fcb2 = (const float*)d_in[23];
  const float* vocab = (const float*)d_in[24];

  char* ws = (char*)d_ws;
  size_t off = 0;
  auto alloc = [&](size_t bytes) -> char* {
    size_t o = (off + 255) & ~(size_t)255; off = o + bytes; return ws + o;
  };
  float* rnorm   = (float*)alloc((size_t)VSZ * 4);
  float* comb    = (float*)alloc((size_t)BQ * 3072 * 4);
  float* h1      = (float*)alloc((size_t)BQ * 4500 * 4);
  float* h2      = (float*)alloc((size_t)BQ * 2000 * 4);
  float* inter   = (float*)alloc((size_t)BQ * 1250 * 4);
  float* heads   = (float*)alloc((size_t)BQ * 3 * DIM * 4);
  float* pooled  = (float*)alloc((size_t)BQ * DIM * 4);
  float* Cbv     = (float*)alloc((size_t)BQ * DIM * 4);
  int*   idx80   = (int*)  alloc((size_t)BQ * KC * 4);
  float* sc80    = (float*)alloc((size_t)BQ * KC * 4);
  float* qh      = (float*)alloc((size_t)BQ * 3 * 812 * 4);
  float* queries = (float*)alloc((size_t)BQ * 3 * DIM * 4);
  float* xbuf    = (float*)alloc((size_t)BQ * DIM * 4);
  float* fch     = (float*)alloc((size_t)BQ * 812 * 4);
  int*   idx5    = (int*)  alloc((size_t)BQ * 5 * 4);
  // reused big region: sims (B*V) -> v_in/v_h -> values -> sims2
  char* big = alloc((size_t)(BQ * KC) * DIM * 4 + (size_t)(BQ * KC) * 812 * 4);
  float* sims   = (float*)big;
  float* v_in   = (float*)big;
  float* v_h    = (float*)(big + (size_t)(BQ * KC) * DIM * 4);
  float* values = (float*)big;

  auto g = [](int n, int m, int tn, int tm) {
    return dim3((unsigned)((n + tn - 1) / tn), (unsigned)((m + tm - 1) / tm));
  };

  // encoder (128x64 tiles, 256 threads)
  pool_concat_kernel<<<BQ, 256, 0, stream>>>(pos, neg, neut, assas, comb);
  gemm_wmma_kernel<8, 4><<<g(4500, BQ, 64, 128), 256, 0, stream>>>(comb, 3072, eW1, 4500, nullptr, eb1, h1, 4500, 0, BQ, 4500, 3072, 0, 1);
  gemm_wmma_kernel<8, 4><<<g(2000, BQ, 64, 128), 256, 0, stream>>>(h1, 4500, eW2, 2000, nullptr, eb2, h2, 2000, 0, BQ, 2000, 4500, 0, 1);
  gemm_wmma_kernel<8, 4><<<g(1250, BQ, 64, 128), 256, 0, stream>>>(h2, 2000, eW3, 1250, nullptr, eb3, inter, 1250, 0, BQ, 1250, 2000, 0, 0);
  for (int h = 0; h < 3; ++h)
    gemm_wmma_kernel<8, 4><<<g(DIM, BQ, 64, 128), 256, 0, stream>>>(inter, 1250, eWh + (size_t)h * 1250 * DIM, DIM,
                                                                    nullptr, ebh + h * DIM, heads, 3 * DIM, h * DIM,
                                                                    BQ, DIM, 1250, 0, 0);
  rownorm_kernel<<<(BQ * 3 + 7) / 8, 256, 0, stream>>>(heads, heads, BQ * 3);   // heads == tri_n
  pooled_kernel<<<(BQ + 7) / 8, 256, 0, stream>>>(heads, pooled);

  // vocab norms + sim1 + top-80  (512x64 tile: vocab streamed exactly once)
  vrnorm_kernel<<<(VSZ + 7) / 8, 256, 0, stream>>>(vocab, rnorm);
  gemm_wmma_kernel<32, 4><<<g(VSZ, BQ, 64, 512), 1024, 0, stream>>>(pooled, DIM, vocab, DIM, rnorm, nullptr, sims, VSZ, 0, BQ, VSZ, DIM, 1, 0);
  topk_hist_kernel<<<BQ, 256, 0, stream>>>(sims, VSZ, KC, idx80, VSZ);

  // rerank scores
  combo_kernel<<<BQ, 256, 0, stream>>>(pos, neut, assas, Cbv);
  scores_kernel<<<BQ, 256, 0, stream>>>(vocab, rnorm, Cbv, idx80, sc80);

  // queries MLP
  gemm_wmma_kernel<8, 4><<<g(812, BQ * 3, 64, 128), 256, 0, stream>>>(heads, DIM, qgW1, 812, nullptr, qgb1, qh, 812, 0, BQ * 3, 812, DIM, 0, 2);
  gemm_wmma_kernel<8, 4><<<g(DIM, BQ * 3, 64, 128), 256, 0, stream>>>(qh, 812, qgW2, DIM, nullptr, qgb2, queries, DIM, 0, BQ * 3, DIM, 812, 0, 0);

  // values MLP (reuses sims region; topk already consumed it)
  vgather_kernel<<<BQ * KC, 256, 0, stream>>>(vocab, idx80, sc80, v_in);
  gemm_wmma_kernel<8, 4><<<g(812, BQ * KC, 64, 128), 256, 0, stream>>>(v_in, DIM, vgW1, 812, nullptr, vgb1, v_h, 812, 0, BQ * KC, 812, DIM, 0, 2);
  gemm_wmma_kernel<8, 4><<<g(DIM, BQ * KC, 64, 128), 256, 0, stream>>>(v_h, 812, vgW2, DIM, nullptr, vgb2, values, DIM, 0, BQ * KC, DIM, 812, 0, 0);

  // attention + residual + norm, then fc MLP + norm -> out
  attn_kernel<<<BQ, 256, 0, stream>>>(queries, vocab, idx80, values, pooled, xbuf);
  gemm_wmma_kernel<8, 4><<<g(812, BQ, 64, 128), 256, 0, stream>>>(xbuf, DIM, fcW1, 812, nullptr, fcb1, fch, 812, 0, BQ, 812, DIM, 0, 2);
  float* outpre = queries;  // queries dead after attn
  gemm_wmma_kernel<8, 4><<<g(DIM, BQ, 64, 128), 256, 0, stream>>>(fch, 812, fcW2, DIM, nullptr, fcb2, outpre, DIM, 0, BQ, DIM, 812, 0, 0);
  float* out0 = (float*)d_out;
  rownorm_kernel<<<(BQ + 7) / 8, 256, 0, stream>>>(outpre, out0, BQ);

  // high = mean of top-5 scored word embeddings
  high_kernel<<<BQ, 256, 0, stream>>>(vocab, idx80, sc80, out0 + (size_t)BQ * DIM);

  // sim2 on out, top-5, gather search embeddings
  gemm_wmma_kernel<32, 4><<<g(VSZ, BQ, 64, 512), 1024, 0, stream>>>(out0, DIM, vocab, DIM, rnorm, nullptr, sims, VSZ, 0, BQ, VSZ, DIM, 1, 0);
  topk_hist_kernel<<<BQ, 256, 0, stream>>>(sims, VSZ, 5, idx5, VSZ);
  sgather_kernel<<<BQ * 5, 256, 0, stream>>>(vocab, idx5, out0 + (size_t)BQ * DIM * 2);
}
